// MogrifierLanguageModel_50680614093413
// MI455X (gfx1250) — matrix operand order repaired and checked by reading, not compiled, not run on hardware
//
#include <hip/hip_runtime.h>
#include <hip/hip_bf16.h>

// Mogrifier LSTM LM for MI455X (gfx1250, wave32, WMMA).
// bf16 v_wmma_f32_16x16x32_bf16 for every GEMM (fp32 accumulate),
// fp32 for all recurrent state / gates / softmax.
// Workspace use: ~56 MB (bf16 weights + hs buffer + states + gates).

#define B_    32
#define S_    128
#define E_    512
#define H_    512
#define V_    32000
#define KR    90
#define KRP   96          // rank padded to multiple of 32 for WMMA K
#define STEPS_ 5
#define G4H   2048        // 4*H gates

typedef __attribute__((ext_vector_type(16))) __bf16 v16bf;
typedef __attribute__((ext_vector_type(8)))  float  v8f;
typedef __attribute__((ext_vector_type(4)))  unsigned int u32x4;

union BF16Tile {
  u32x4 q[2];
  unsigned int u[8];
  unsigned short s[16];
  v16bf v;
};

__device__ __forceinline__ unsigned short f2bf(float f) {
  unsigned int u = __float_as_uint(f);
  u += 0x7FFFu + ((u >> 16) & 1u);            // round-to-nearest-even
  return (unsigned short)(u >> 16);
}

__device__ __forceinline__ float sigm(float x) { return 1.0f / (1.0f + expf(-x)); }

// B-matrix tile fragment: 16 contiguous bf16 starting at p.
// Layout used here (contiguous K, not the ISA table order): lane%16 = N, lane/16 picks K-half of 16,
// VGPR v holds K = half*16 + 2v, 2v+1  -> contiguous 16 values.
__device__ __forceinline__ v16bf ld_b_tile(const unsigned short* p) {
  BF16Tile t;
  t.q[0] = *(const u32x4*)(p);
  t.q[1] = *(const u32x4*)(p + 8);
  return t.v;
}

// A-matrix tile fragment: 16 contiguous bf16 starting at p (K = half*16 + 0..15).
// Layout: lane%16 = M, half = lane/16; same contiguous K order as B.
// VGPR v: K = half*16 + 2v, 2v+1  -> contiguous 16 values.
__device__ __forceinline__ v16bf ld_a_tile(const unsigned short* p) {
  BF16Tile t;
  t.q[0] = *(const u32x4*)(p);
  t.q[1] = *(const u32x4*)(p + 8);
  return t.v;
}

__device__ __forceinline__ v8f wmma_bf16(v16bf a, v16bf b, v8f c) {
  return __builtin_amdgcn_wmma_f32_16x16x32_bf16(false, a, false, b, (short)0, c, false, false);
}

// ---------------------------------------------------------------- prep kernels

__global__ void cvt_bf16_kernel(const float* __restrict__ src,
                                unsigned short* __restrict__ dst, int n) {
  int i = blockIdx.x * blockDim.x + threadIdx.x;
  if (i < n) dst[i] = f2bf(src[i]);
}

// U: [STEPS][H][KR] -> Ut: [STEPS][KRP][H] (transposed, zero padded rows KR..KRP-1)
__global__ void cvt_transpose_U_kernel(const float* __restrict__ U,
                                       unsigned short* __restrict__ Ut) {
  int i = blockIdx.x * blockDim.x + threadIdx.x;
  if (i >= STEPS_ * KRP * H_) return;
  int st = i / (KRP * H_);
  int r  = i - st * KRP * H_;
  int j  = r / H_;
  int k  = r - j * H_;
  Ut[i] = (j < KR) ? f2bf(U[(size_t)st * H_ * KR + (size_t)k * KR + j]) : (unsigned short)0;
}

// V: [STEPS][KR][H] -> Vt: [STEPS][H][KRP] (transposed, zero padded cols KR..KRP-1)
__global__ void cvt_transpose_V_kernel(const float* __restrict__ Vm,
                                       unsigned short* __restrict__ Vt) {
  int i = blockIdx.x * blockDim.x + threadIdx.x;
  if (i >= STEPS_ * H_ * KRP) return;
  int st = i / (H_ * KRP);
  int r  = i - st * H_ * KRP;
  int n  = r / KRP;
  int j  = r - n * KRP;
  Vt[i] = (j < KR) ? f2bf(Vm[(size_t)st * KR * H_ + (size_t)j * H_ + n]) : (unsigned short)0;
}

__global__ void zero_kernel(float* __restrict__ p, int n) {
  int i = blockIdx.x * blockDim.x + threadIdx.x;
  if (i < n) p[i] = 0.0f;
}

// ---------------------------------------------------------------- mogrifier

// Padded LDS strides (distinct banks across the 16 rows of a tile).
constexpr int XS  = 516;   // f32 row stride for X/H   (516 % 64 == 4)
constexpr int ABS = 520;   // bf16 row stride, packed input  (1040B, 16B aligned)
constexpr int TBS = 104;   // bf16 row stride, rank temp     (208B, 16B aligned)
constexpr int MOG_LDS_BYTES =
    2 * B_ * XS * (int)sizeof(float) + (B_ * ABS + B_ * TBS) * (int)sizeof(short);

// One workgroup (16 waves) runs all 5 alternating low-rank gated updates.
// x,h live in LDS fp32; per step the input matrix is packed to bf16 once,
// both GEMMs feed WMMA from b128 LDS fragment loads.
__global__ __launch_bounds__(512) void mogrify_kernel(
    const float* __restrict__ xsrc, const int* __restrict__ idx, int t,
    const float* __restrict__ hstate,
    const unsigned short* __restrict__ Ut,   // [STEPS][KRP][H]
    const unsigned short* __restrict__ Vt,   // [STEPS][H][KRP]
    const float* __restrict__ bm,            // [STEPS][H]
    unsigned short* __restrict__ xout,       // [B][H] bf16
    unsigned short* __restrict__ hout)       // [B][H] bf16
{
  extern __shared__ char smem[];
  float* X  = (float*)smem;                       // [B][XS] fp32
  float* Hd = X + B_ * XS;                        // [B][XS] fp32
  unsigned short* Ab = (unsigned short*)(Hd + B_ * XS); // [B][ABS] bf16 packed input
  unsigned short* Tb = Ab + B_ * ABS;             // [B][TBS] bf16 rank temp

  const int tid  = threadIdx.x;
  const int lane = tid & 31;
  const int wave = tid >> 5;
  const int hf   = (lane >> 4) & 1;
  const int l15  = lane & 15;

  for (int j = tid; j < B_ * H_; j += 512) {
    int m = j >> 9, e = j & 511;
    const float* row = idx ? (xsrc + (size_t)idx[m * S_ + t] * E_)
                           : (xsrc + (size_t)m * E_);
    X[m * XS + e]  = row[e];
    Hd[m * XS + e] = hstate[j];
  }
  __syncthreads();

  for (int st = 0; st < STEPS_; ++st) {
    // even st: input = h, target = x   (reference: i+1 odd -> update x)
    const float* inp = (st & 1) ? X : Hd;
    float*       tgt = (st & 1) ? Hd : X;
    const unsigned short* U  = Ut + (size_t)st * KRP * H_;
    const unsigned short* Vw = Vt + (size_t)st * H_ * KRP;
    const float*          bb = bm + st * H_;

    // pack input matrix to bf16 once per step
    for (int j = tid; j < B_ * H_; j += 512) {
      int m = j >> 9, e = j & 511;
      Ab[m * ABS + e] = f2bf(inp[m * XS + e]);
    }
    __syncthreads();

    // GEMM1: Tb[32][96] = inp(32x512) @ U^T   (2 x 6 tiles -> waves 0..11)
    if (wave < 12) {
      const int mtile = wave & 1, ntile = wave >> 1;
      const int Mr = mtile * 16 + l15;
      const int Nr = ntile * 16 + l15;
      v8f acc = {};
      for (int kk = 0; kk < 16; ++kk) {
        v16bf a = ld_a_tile(Ab + Mr * ABS + kk * 32 + hf * 16);
        v16bf b = ld_b_tile(U + (size_t)Nr * H_ + kk * 32 + hf * 16);
        acc = wmma_bf16(a, b, acc);
      }
#pragma unroll
      for (int r = 0; r < 8; ++r) {
        int m = mtile * 16 + r + hf * 8;
        Tb[m * TBS + Nr] = f2bf(acc[r]);
      }
    }
    __syncthreads();

    // GEMM2: gate(32x512) = Tb(32x96) @ V ; 2 x 32 = 64 tiles, 4 per wave
    for (int tt = wave; tt < 64; tt += 16) {
      const int mtile = tt & 1, ntile = tt >> 1;
      const int Mr = mtile * 16 + l15;
      const int Nr = ntile * 16 + l15;
      v8f acc = {};
#pragma unroll
      for (int kk = 0; kk < 3; ++kk) {
        v16bf a = ld_a_tile(Tb + Mr * TBS + kk * 32 + hf * 16);
        v16bf b = ld_b_tile(Vw + (size_t)Nr * KRP + kk * 32 + hf * 16);
        acc = wmma_bf16(a, b, acc);
      }
      float bn = bb[Nr];
#pragma unroll
      for (int r = 0; r < 8; ++r) {
        int m = mtile * 16 + r + hf * 8;
        float g = 2.0f * sigm(acc[r] + bn);
        tgt[m * XS + Nr] *= g;
      }
    }
    __syncthreads();
  }

  for (int j = tid; j < B_ * H_; j += 512) {
    int m = j >> 9, e = j & 511;
    xout[j] = f2bf(X[m * XS + e]);
    hout[j] = f2bf(Hd[m * XS + e]);
  }
}

// ---------------------------------------------------------------- LSTM

// gates[32][2048] = x@Wih^T + h@Whh^T + bih + bhh.
// 2-way N blocking: each wave does 1 M-tile x 2 N-tiles, A fragments reused.
__global__ __launch_bounds__(256) void lstm_gates_kernel(
    const unsigned short* __restrict__ xb, const unsigned short* __restrict__ hb,
    const unsigned short* __restrict__ Wih, const unsigned short* __restrict__ Whh,
    const float* __restrict__ bih, const float* __restrict__ bhh,
    float* __restrict__ gates)
{
  const int lane  = threadIdx.x & 31;
  const int gw    = blockIdx.x * 8 + (threadIdx.x >> 5);  // 0..127
  const int mtile = gw & 1, npair = gw >> 1;              // 2 x 64 groups
  const int hf = (lane >> 4) & 1, l15 = lane & 15;
  const int Mr  = mtile * 16 + l15;
  const int Nr0 = (npair * 2) * 16 + l15;
  const int Nr1 = Nr0 + 16;
  v8f acc0 = {}, acc1 = {};
  for (int kk = 0; kk < 16; ++kk) {
    v16bf a1  = ld_a_tile(xb + (size_t)Mr * H_ + kk * 32 + hf * 16);
    v16bf b00 = ld_b_tile(Wih + (size_t)Nr0 * H_ + kk * 32 + hf * 16);
    v16bf b01 = ld_b_tile(Wih + (size_t)Nr1 * H_ + kk * 32 + hf * 16);
    acc0 = wmma_bf16(a1, b00, acc0);
    acc1 = wmma_bf16(a1, b01, acc1);
    v16bf a2  = ld_a_tile(hb + (size_t)Mr * H_ + kk * 32 + hf * 16);
    v16bf b10 = ld_b_tile(Whh + (size_t)Nr0 * H_ + kk * 32 + hf * 16);
    v16bf b11 = ld_b_tile(Whh + (size_t)Nr1 * H_ + kk * 32 + hf * 16);
    acc0 = wmma_bf16(a2, b10, acc0);
    acc1 = wmma_bf16(a2, b11, acc1);
  }
  float bias0 = bih[Nr0] + bhh[Nr0];
  float bias1 = bih[Nr1] + bhh[Nr1];
#pragma unroll
  for (int r = 0; r < 8; ++r) {
    int m = mtile * 16 + r + hf * 8;   // C/D layout: VGPR r -> M = r + half*8
    gates[(size_t)m * G4H + Nr0] = acc0[r] + bias0;
    gates[(size_t)m * G4H + Nr1] = acc1[r] + bias1;
  }
}

// pointwise cell: c' = sig(f)*c + sig(i)*tanh(g); h' = sig(o)*tanh(c')
__global__ __launch_bounds__(256) void lstm_point_kernel(
    const float* __restrict__ gates,
    float* __restrict__ hstate, float* __restrict__ cstate,
    float* __restrict__ hsout, int t)
{
  int j = blockIdx.x * blockDim.x + threadIdx.x;   // exactly B*H threads
  int m = j >> 9, e = j & 511;
  const float* gr = gates + (size_t)m * G4H;
  float ig = sigm(gr[e]);
  float fg = sigm(gr[512 + e]);
  float gg = tanhf(gr[1024 + e]);
  float og = sigm(gr[1536 + e]);
  float c  = fg * cstate[j] + ig * gg;
  float h  = og * tanhf(c);
  cstate[j] = c;
  hstate[j] = h;
  if (hsout) hsout[(size_t)m * S_ * H_ + (size_t)t * H_ + e] = h;
}

// ---------------------------------------------------------------- output head

// logits[4096][32000] = hs @ Wout^T + bout.
// 4-way N blocking: each wave does 1 M-tile x 4 N-tiles; A fragment feeds 4 WMMAs.
// 8 waves of a block share the N-group -> B tiles hit L0/L2.
__global__ __launch_bounds__(256) void out_gemm_kernel(
    const unsigned short* __restrict__ hsb, const unsigned short* __restrict__ Wout,
    const float* __restrict__ bout, float* __restrict__ out)
{
  const int lane  = threadIdx.x & 31;
  const int gw    = blockIdx.x * 8 + (threadIdx.x >> 5);  // 0..127999
  const int mtile = gw & 255;                             // 0..255
  const int ngrp  = gw >> 8;                              // 0..499
  const int hf = (lane >> 4) & 1, l15 = lane & 15;
  const int Mr  = mtile * 16 + l15;
  const int Nr0 = (ngrp * 4) * 16 + l15;
  v8f acc0 = {}, acc1 = {}, acc2 = {}, acc3 = {};
  for (int kk = 0; kk < 16; ++kk) {
    v16bf a  = ld_a_tile(hsb + (size_t)Mr * H_ + kk * 32 + hf * 16);
    const unsigned short* wp = Wout + kk * 32 + hf * 16;
    v16bf b0 = ld_b_tile(wp + (size_t)(Nr0)      * H_);
    v16bf b1 = ld_b_tile(wp + (size_t)(Nr0 + 16) * H_);
    v16bf b2 = ld_b_tile(wp + (size_t)(Nr0 + 32) * H_);
    v16bf b3 = ld_b_tile(wp + (size_t)(Nr0 + 48) * H_);
    __builtin_prefetch(wp + (size_t)Nr0 * H_ + 32, 0, 3);   // stream next K chunk
    acc0 = wmma_bf16(a, b0, acc0);
    acc1 = wmma_bf16(a, b1, acc1);
    acc2 = wmma_bf16(a, b2, acc2);
    acc3 = wmma_bf16(a, b3, acc3);
  }
#pragma unroll
  for (int r = 0; r < 8; ++r) {
    int m = mtile * 16 + r + hf * 8;
    float* po = out + (size_t)m * V_;
    po[Nr0]      = acc0[r] + bout[Nr0];
    po[Nr0 + 16] = acc1[r] + bout[Nr0 + 16];
    po[Nr0 + 32] = acc2[r] + bout[Nr0 + 32];
    po[Nr0 + 48] = acc3[r] + bout[Nr0 + 48];
  }
}

// in-place log_softmax over V per row: one block per row, online logsumexp.
__global__ __launch_bounds__(256) void log_softmax_kernel(float* __restrict__ out)
{
  __shared__ float sm[256], ss[256];
  float* p = out + (size_t)blockIdx.x * V_;
  int tid = threadIdx.x;
  float m = -__builtin_inff(), s = 0.0f;
  for (int v = tid; v < V_; v += 256) {
    float x = p[v];
    if (x > m) { s = s * expf(m - x) + 1.0f; m = x; }
    else        s += expf(x - m);
  }
  sm[tid] = m; ss[tid] = s;
  __syncthreads();
  for (int off = 128; off > 0; off >>= 1) {
    if (tid < off) {
      float m1 = sm[tid], s1 = ss[tid];
      float m2 = sm[tid + off], s2 = ss[tid + off];
      float mm = fmaxf(m1, m2);
      sm[tid] = mm;
      ss[tid] = s1 * expf(m1 - mm) + s2 * expf(m2 - mm);
    }
    __syncthreads();
  }
  float L = sm[0] + logf(ss[0]);
  for (int v = tid; v < V_; v += 256) p[v] -= L;
}

// ---------------------------------------------------------------- host

extern "C" void kernel_launch(void* const* d_in, const int* in_sizes, int n_in,
                              void* d_out, int out_size, void* d_ws, size_t ws_size,
                              hipStream_t stream)
{
  (void)in_sizes; (void)n_in; (void)out_size; (void)ws_size;
  const int*   inputs = (const int*)  d_in[0];
  const float* emb    = (const float*)d_in[1];
  const float* W_ih1  = (const float*)d_in[2];
  const float* W_hh1  = (const float*)d_in[3];
  const float* b_ih1  = (const float*)d_in[4];
  const float* b_hh1  = (const float*)d_in[5];
  const float* mog1_U = (const float*)d_in[6];
  const float* mog1_V = (const float*)d_in[7];
  const float* mog1_b = (const float*)d_in[8];
  const float* W_ih2  = (const float*)d_in[9];
  const float* W_hh2  = (const float*)d_in[10];
  const float* b_ih2  = (const float*)d_in[11];
  const float* b_hh2  = (const float*)d_in[12];
  const float* mog2_U = (const float*)d_in[13];
  const float* mog2_V = (const float*)d_in[14];
  const float* mog2_b = (const float*)d_in[15];
  const float* W_out  = (const float*)d_in[16];
  const float* b_out  = (const float*)d_in[17];
  float* out = (float*)d_out;

  // workspace carve-up (~56 MB total)
  char* ws = (char*)d_ws;
  size_t off = 0;
  auto carve = [&](size_t bytes) -> void* {
    void* p = ws + off;
    off = (off + bytes + 255) & ~(size_t)255;
    return p;
  };
  unsigned short* Wih1b = (unsigned short*)carve((size_t)G4H * H_ * 2);
  unsigned short* Whh1b = (unsigned short*)carve((size_t)G4H * H_ * 2);
  unsigned short* Wih2b = (unsigned short*)carve((size_t)G4H * H_ * 2);
  unsigned short* Whh2b = (unsigned short*)carve((size_t)G4H * H_ * 2);
  unsigned short* Woutb = (unsigned short*)carve((size_t)V_ * H_ * 2);
  unsigned short* Ut1   = (unsigned short*)carve((size_t)STEPS_ * KRP * H_ * 2);
  unsigned short* Vt1   = (unsigned short*)carve((size_t)STEPS_ * H_ * KRP * 2);
  unsigned short* Ut2   = (unsigned short*)carve((size_t)STEPS_ * KRP * H_ * 2);
  unsigned short* Vt2   = (unsigned short*)carve((size_t)STEPS_ * H_ * KRP * 2);
  float* hsf            = (float*)carve((size_t)B_ * S_ * H_ * 4);
  unsigned short* hsb   = (unsigned short*)carve((size_t)B_ * S_ * H_ * 2);
  float* h1 = (float*)carve((size_t)B_ * H_ * 4);
  float* c1 = (float*)carve((size_t)B_ * H_ * 4);
  float* h2 = (float*)carve((size_t)B_ * H_ * 4);
  float* c2 = (float*)carve((size_t)B_ * H_ * 4);
  unsigned short* x1b = (unsigned short*)carve((size_t)B_ * H_ * 2);
  unsigned short* hmb = (unsigned short*)carve((size_t)B_ * H_ * 2);
  float* gates = (float*)carve((size_t)B_ * G4H * 4);

  // allow >64KB dynamic LDS for the fused mogrifier kernel (320KB/WGP on CDNA5)
  hipFuncSetAttribute((const void*)mogrify_kernel,
                      hipFuncAttributeMaxDynamicSharedMemorySize, MOG_LDS_BYTES);

  // ---- weight prep (runs every call; deterministic)
  const int nW = G4H * H_;                         // 1,048,576
  cvt_bf16_kernel<<<(nW + 255) / 256, 256, 0, stream>>>(W_ih1, Wih1b, nW);
  cvt_bf16_kernel<<<(nW + 255) / 256, 256, 0, stream>>>(W_hh1, Whh1b, nW);
  cvt_bf16_kernel<<<(nW + 255) / 256, 256, 0, stream>>>(W_ih2, Wih2b, nW);
  cvt_bf16_kernel<<<(nW + 255) / 256, 256, 0, stream>>>(W_hh2, Whh2b, nW);
  const int nWo = V_ * H_;                         // 16,384,000
  cvt_bf16_kernel<<<(nWo + 255) / 256, 256, 0, stream>>>(W_out, Woutb, nWo);
  const int nUt = STEPS_ * KRP * H_;               // 245,760
  cvt_transpose_U_kernel<<<(nUt + 255) / 256, 256, 0, stream>>>(mog1_U, Ut1);
  cvt_transpose_V_kernel<<<(nUt + 255) / 256, 256, 0, stream>>>(mog1_V, Vt1);
  cvt_transpose_U_kernel<<<(nUt + 255) / 256, 256, 0, stream>>>(mog2_U, Ut2);
  cvt_transpose_V_kernel<<<(nUt + 255) / 256, 256, 0, stream>>>(mog2_V, Vt2);
  const int nS = B_ * H_;
  zero_kernel<<<(nS + 255) / 256, 256, 0, stream>>>(h1, nS);
  zero_kernel<<<(nS + 255) / 256, 256, 0, stream>>>(c1, nS);
  zero_kernel<<<(nS + 255) / 256, 256, 0, stream>>>(h2, nS);
  zero_kernel<<<(nS + 255) / 256, 256, 0, stream>>>(c2, nS);

  // ---- recurrent scan (sequential in t; stream order gives dependencies)
  for (int t = 0; t < S_; ++t) {
    // layer 1: x from embedding gather, h from h1 state
    mogrify_kernel<<<1, 512, MOG_LDS_BYTES, stream>>>(
        emb, inputs, t, h1, Ut1, Vt1, mog1_b, x1b, hmb);
    lstm_gates_kernel<<<16, 256, 0, stream>>>(
        x1b, hmb, Wih1b, Whh1b, b_ih1, b_hh1, gates);
    lstm_point_kernel<<<(B_ * H_) / 256, 256, 0, stream>>>(
        gates, h1, c1, (float*)nullptr, t);
    // layer 2: x = new h1, h from h2 state
    mogrify_kernel<<<1, 512, MOG_LDS_BYTES, stream>>>(
        h1, (const int*)nullptr, t, h2, Ut2, Vt2, mog2_b, x1b, hmb);
    lstm_gates_kernel<<<16, 256, 0, stream>>>(
        x1b, hmb, Wih2b, Whh2b, b_ih2, b_hh2, gates);
    lstm_point_kernel<<<(B_ * H_) / 256, 256, 0, stream>>>(
        gates, h2, c2, hsf, t);
  }

  // ---- output head
  const int nHs = B_ * S_ * H_;                    // 2,097,152
  cvt_bf16_kernel<<<(nHs + 255) / 256, 256, 0, stream>>>(hsf, hsb, nHs);
  // 256 M-tiles x 500 N-groups(4 tiles) = 128000 waves -> 16000 blocks of 8 waves
  out_gemm_kernel<<<16000, 256, 0, stream>>>(hsb, Woutb, b_out, out);
  log_softmax_kernel<<<B_ * S_, 256, 0, stream>>>(out);
}
